// mamba_fusion4_26371099198484
// MI455X (gfx1250) — matrix operand order repaired and checked by reading, not compiled, not run on hardware
//
#include <hip/hip_runtime.h>
#include <hip/hip_bf16.h>

// ---------------------------------------------------------------------------
// VMamba SS2D block for MI455X (gfx1250, wave32, WMMA 16x16x32 f16)
// ---------------------------------------------------------------------------
#define BB   8
#define CC   384
#define HH   32
#define WW2  32
#define LL   1024          // H*W
#define MROWS 8192         // B*L
#define DI   768
#define NN   4
#define RR   24
#define KK   4
#define HID  1536

typedef __attribute__((ext_vector_type(16))) _Float16 v16h;
typedef __attribute__((ext_vector_type(8)))  float    v8f;
typedef unsigned int v4u __attribute__((ext_vector_type(4)));
typedef unsigned int v8u __attribute__((ext_vector_type(8)));

union Frag16 { v16h h; uint4 u[2]; };

enum { GM_PLAIN = 0, GM_XRES = 1, GM_GELU = 2, GM_BIASRES = 3 };

// ---------------------------------------------------------------------------
// Generic WMMA GEMM:  Out[M,N] = A[M,K](f16, row-major) * W[N,K](f16)^T
// Each wave computes a 16 x (16*NT) tile, K-loop in steps of 32, with
// register double-buffering: fragments for step kk+32 are loaded before the
// wmmas for step kk are issued, so load latency overlaps matrix math.
// ---------------------------------------------------------------------------
template <int NT, int MODE>
__global__ __launch_bounds__(256) void gemm_wmma_kernel(
    const _Float16* __restrict__ A, const _Float16* __restrict__ W,
    float* __restrict__ outF, _Float16* __restrict__ outH,
    const float* __restrict__ bias, const float* __restrict__ res,
    int M, int N, int K)
{
    const int wave = (blockIdx.x * blockDim.x + threadIdx.x) >> 5;
    const int lane = threadIdx.x & 31;
    const int tiles_m = M >> 4;
    const int tiles_n = N / (16 * NT);
    if (wave >= tiles_m * tiles_n) return;   // wave-uniform exit: EXEC stays all-ones
    const int mt  = wave % tiles_m;
    const int ntb = (wave / tiles_m) * NT;

    const int nl  = lane & 15;
    const int sel = lane >> 4;

    // A fragment: lane holds row M = mt*16 + nl.
    // halves 0..7  -> K = kk + sel*8 + h ; halves 8..15 -> K = kk + 16 + sel*8 + h
    const _Float16* Ap = A + (size_t)((mt << 4) + nl) * K + sel * 8;
    // B fragment: lane holds col n; halves 0..7 -> K = kk + sel*8 + h ; halves 8..15 -> K = kk + 16 + sel*8 + h
    const _Float16* Wp = W + (size_t)((ntb << 4) + nl) * K + sel * 8;

    v8f acc[NT];
#pragma unroll
    for (int j = 0; j < NT; ++j) { v8f z = {}; acc[j] = z; }

    // prologue: load fragments for kk = 0
    Frag16 a0, b0[NT];
    a0.u[0] = *(const uint4*)(Ap);
    a0.u[1] = *(const uint4*)(Ap + 16);
#pragma unroll
    for (int j = 0; j < NT; ++j) {
        const _Float16* wp = Wp + (size_t)j * 16 * K;
        b0[j].u[0] = *(const uint4*)(wp);
        b0[j].u[1] = *(const uint4*)(wp + 16);
    }

    for (int kk = 32; kk < K; kk += 32) {
        // issue next-step loads first (independent of pending wmmas)
        Frag16 a1, b1[NT];
        a1.u[0] = *(const uint4*)(Ap + kk);
        a1.u[1] = *(const uint4*)(Ap + kk + 16);
#pragma unroll
        for (int j = 0; j < NT; ++j) {
            const _Float16* wp = Wp + (size_t)j * 16 * K + kk;
            b1[j].u[0] = *(const uint4*)(wp);
            b1[j].u[1] = *(const uint4*)(wp + 16);
        }
        // prefetch two K-steps ahead (gfx1250 global_prefetch_b8)
        if (kk + 64 < K) {
            __builtin_prefetch(Ap + kk + 64, 0, 0);
            __builtin_prefetch(Wp + kk + 64, 0, 0);
        }
        // math on current fragments
#pragma unroll
        for (int j = 0; j < NT; ++j)
            acc[j] = __builtin_amdgcn_wmma_f32_16x16x32_f16(
                false, a0.h, false, b0[j].h, (short)0, acc[j], false, false);
        a0 = a1;
#pragma unroll
        for (int j = 0; j < NT; ++j) b0[j] = b1[j];
    }
    // epilogue wmmas
#pragma unroll
    for (int j = 0; j < NT; ++j)
        acc[j] = __builtin_amdgcn_wmma_f32_16x16x32_f16(
            false, a0.h, false, b0[j].h, (short)0, acc[j], false, false);

    // C/D layout: lane -> col n = lane&15 ; VGPR v -> row = mt*16 + sel*8 + v
#pragma unroll
    for (int j = 0; j < NT; ++j) {
        const int ncol = ((ntb + j) << 4) + nl;
#pragma unroll
        for (int v = 0; v < 8; ++v) {
            const int r   = (mt << 4) + (sel << 3) + v;
            const float val = acc[j][v];
            const size_t oi = (size_t)r * N + ncol;
            if constexpr (MODE == GM_PLAIN) {
                outF[oi] = val;
            } else if constexpr (MODE == GM_XRES) {
                // residual = x (B,C,H,W): res[((b*C + ncol)<<10) + hw], N == C
                const int b = r >> 10, hw = r & 1023;
                outF[oi] = val + res[(((size_t)b * CC + ncol) << 10) + hw];
            } else if constexpr (MODE == GM_GELU) {
                const float t = val + bias[ncol];
                const float g = 0.5f * t *
                    (1.f + tanhf(0.7978845608f * (t + 0.044715f * t * t * t)));
                outH[oi] = (_Float16)g;
            } else { // GM_BIASRES
                outF[oi] = val + bias[ncol] + res[oi];
            }
        }
    }
}

// ---------------------------------------------------------------------------
// TDM warm kernel: builds a Tensor DMA descriptor (D# group0 + group1, per
// CDNA5 ISA ch.8) and DMAs a 64x32 f16 tile of the weight matrix into LDS.
// Used purely as an L2 warm for the weight matrices ahead of the GEMMs;
// exercises tensor_load_to_lds + TENSORcnt. One wave per block.
// ---------------------------------------------------------------------------
__global__ __launch_bounds__(32) void tdm_warm_kernel(
    const _Float16* __restrict__ w, int rows, int cols)
{
    __shared__ _Float16 stage[64 * 32];
    const unsigned long long gaddr =
        (unsigned long long)(const void*)(w + (size_t)blockIdx.x * 64 * cols);
    const unsigned int lds = (unsigned int)(unsigned long long)(&stage[0]);

    // D# group 0: count=1 | lds_addr | global_addr[56:0] | type=2
    v4u g0;
    g0[0] = 1u;
    g0[1] = lds;
    g0[2] = (unsigned int)gaddr;
    g0[3] = ((unsigned int)(gaddr >> 32) & 0x01FFFFFFu) | (2u << 30);

    // D# group 1: data_size=1 (2B), tensor dims (cols x rows), tile 32x64,
    // dim0 stride = cols. Field packing per ISA 8.4 (bits 255:0).
    v8u g1;
    g1[0] = 1u << 16;                                              // data_size
    g1[1] = ((unsigned int)cols & 0xFFFFu) << 16;                  // dim0 lo16
    g1[2] = (((unsigned int)cols >> 16) & 0xFFFFu) |
            (((unsigned int)rows & 0xFFFFu) << 16);                // dim0 hi | dim1 lo
    g1[3] = (((unsigned int)rows >> 16) & 0xFFFFu) | (32u << 16);  // dim1 hi | tile0
    g1[4] = 64u;                                                   // tile1 (tile2=0)
    g1[5] = (unsigned int)cols;                                    // dim0 stride lo32
    g1[6] = 0u;                                                    // stride hi | d1 stride lo
    g1[7] = 0u;

    asm volatile("tensor_load_to_lds %0, %1" :: "s"(g0), "s"(g1) : "memory");
    __builtin_amdgcn_s_wait_tensorcnt(0);
}

// ---------------------------------------------------------------------------
__device__ inline float waveRed(float v)
{
#pragma unroll
    for (int m = 16; m >= 1; m >>= 1) v += __shfl_xor(v, m, 32);
    return v;
}

// LN1: per-pixel LayerNorm over C, reading x in (B,C,H,W); f16 out (m, C)
__global__ __launch_bounds__(256) void ln1_kernel(
    const float* __restrict__ x, const float* __restrict__ g,
    const float* __restrict__ bt, _Float16* __restrict__ out)
{
    const int wave = (blockIdx.x * blockDim.x + threadIdx.x) >> 5;
    const int lane = threadIdx.x & 31;
    if (wave >= MROWS) return;
    const int b = wave >> 10, hw = wave & 1023;
    const float* xp = x + ((size_t)b * CC) * LL + hw;
    float vals[12], s = 0.f, ss = 0.f;
#pragma unroll
    for (int i = 0; i < 12; ++i) {
        const int c = lane + 32 * i;
        const float v = xp[(size_t)c * LL];
        vals[i] = v; s += v; ss += v * v;
    }
    s = waveRed(s); ss = waveRed(ss);
    const float mu = s * (1.f / CC);
    const float rs = rsqrtf(ss * (1.f / CC) - mu * mu + 1e-5f);
#pragma unroll
    for (int i = 0; i < 12; ++i) {
        const int c = lane + 32 * i;
        out[(size_t)wave * CC + c] = (_Float16)((vals[i] - mu) * rs * g[c] + bt[c]);
    }
}

// LN2: contiguous rows (m, C) f32 -> f16
__global__ __launch_bounds__(256) void ln_rows_kernel(
    const float* __restrict__ in, const float* __restrict__ g,
    const float* __restrict__ bt, _Float16* __restrict__ out)
{
    const int wave = (blockIdx.x * blockDim.x + threadIdx.x) >> 5;
    const int lane = threadIdx.x & 31;
    if (wave >= MROWS) return;
    const float* p = in + (size_t)wave * CC;
    float vals[12], s = 0.f, ss = 0.f;
#pragma unroll
    for (int i = 0; i < 12; ++i) {
        const float v = p[lane + 32 * i];
        vals[i] = v; s += v; ss += v * v;
    }
    s = waveRed(s); ss = waveRed(ss);
    const float mu = s * (1.f / CC);
    const float rs = rsqrtf(ss * (1.f / CC) - mu * mu + 1e-5f);
#pragma unroll
    for (int i = 0; i < 12; ++i) {
        const int c = lane + 32 * i;
        out[(size_t)wave * CC + c] = (_Float16)((vals[i] - mu) * rs * g[c] + bt[c]);
    }
}

// out-norm over DI=768 then * silu(z); z lives in xz[:, DI:2*DI]
__global__ __launch_bounds__(256) void ynorm_kernel(
    const float* __restrict__ y, const float* __restrict__ xz,
    const float* __restrict__ g, const float* __restrict__ bt,
    _Float16* __restrict__ out)
{
    const int wave = (blockIdx.x * blockDim.x + threadIdx.x) >> 5;
    const int lane = threadIdx.x & 31;
    if (wave >= MROWS) return;
    const float* yp = y + (size_t)wave * DI;
    float vals[24], s = 0.f, ss = 0.f;
#pragma unroll
    for (int i = 0; i < 24; ++i) {
        const float v = yp[lane + 32 * i];
        vals[i] = v; s += v; ss += v * v;
    }
    s = waveRed(s); ss = waveRed(ss);
    const float mu = s * (1.f / DI);
    const float rs = rsqrtf(ss * (1.f / DI) - mu * mu + 1e-5f);
    const float* zp = xz + (size_t)wave * (2 * DI) + DI;
#pragma unroll
    for (int i = 0; i < 24; ++i) {
        const int c = lane + 32 * i;
        const float z  = zp[c];
        const float sz = z / (1.f + __expf(-z));
        out[(size_t)wave * DI + c] =
            (_Float16)(((vals[i] - mu) * rs * g[c] + bt[c]) * sz);
    }
}

// depthwise 3x3 conv + bias + SiLU ; input xz[:, :DI] rows (b*L+hw)
__global__ __launch_bounds__(256) void conv_silu_kernel(
    const float* __restrict__ xz, const float* __restrict__ cw,
    const float* __restrict__ cb, float* __restrict__ xc2)
{
    const int d = blockIdx.x * 256 + threadIdx.x;   // gridDim.x == 3 -> d < 768
    const int m = blockIdx.y;                       // 0..8191
    const int b = m >> 10, hw = m & 1023;
    const int h = hw >> 5, w = hw & 31;
    float acc = cb[d];
#pragma unroll
    for (int i = 0; i < 3; ++i) {
        const int hh = h + i - 1;
        if (hh < 0 || hh > 31) continue;
#pragma unroll
        for (int j = 0; j < 3; ++j) {
            const int ww = w + j - 1;
            if (ww < 0 || ww > 31) continue;
            acc += xz[((size_t)((b << 10) | (hh << 5) | ww)) * (2 * DI) + d] *
                   cw[d * 9 + i * 3 + j];
        }
    }
    acc = acc / (1.f + __expf(-acc));               // SiLU
    xc2[(size_t)m * DI + d] = acc;
}

__device__ inline int dir_hw(int k, int l)
{
    if (k == 0) return l;
    if (k == 1) return ((l & 31) << 5) | (l >> 5);
    if (k == 2) return 1023 - l;
    const int lr = 1023 - l;
    return ((lr & 31) << 5) | (lr >> 5);
}

// build directional f16 sequences xs16[(k*8+b)*L + l][d] from xc2[(b*L+hw)][d]
__global__ __launch_bounds__(256) void xs_build_kernel(
    const float* __restrict__ xc2, _Float16* __restrict__ xs16)
{
    const size_t idx = (size_t)blockIdx.x * 256 + threadIdx.x; // exact cover
    const int d = (int)(idx % DI);
    const size_t r = idx / DI;
    const int l = (int)(r % LL);
    const int kb = (int)(r / LL);
    const int b = kb & 7, k = kb >> 3;
    const int hw = dir_hw(k, l);
    xs16[idx] = (_Float16)xc2[((size_t)((b << 10) + hw)) * DI + d];
}

// ---------------------------------------------------------------------------
// Fused selective scan: thread = (b,k,d). dt-projection (R=24) recomputed in
// registers from the LDS-broadcast x_dbl row; 4-state recurrence in registers;
// directional recombination fused via one relaxed f32 atomic add per step.
// ---------------------------------------------------------------------------
__global__ __launch_bounds__(256) void scan_kernel(
    const float* __restrict__ xdbl,   // (k, b*L+l, 32)
    const float* __restrict__ xc2,    // (b*L+hw, DI)
    const float* __restrict__ dtW,    // (k, DI, R)
    const float* __restrict__ dtB,    // (k, DI)
    const float* __restrict__ Alog,   // (k, DI, N)
    const float* __restrict__ Dvec,   // (k, DI)
    float* __restrict__ ybuf)         // (b*L+hw, DI) accumulated
{
    __shared__ float sh[32];
    const int tid = threadIdx.x;
    const int d   = blockIdx.x * 256 + tid;
    const int bk  = blockIdx.y;
    const int k   = bk >> 3, b = bk & 7;

    float wreg[RR];
    const float* wp = dtW + ((size_t)k * DI + d) * RR;
#pragma unroll
    for (int r = 0; r < RR; ++r) wreg[r] = wp[r];
    const float dtb = dtB[k * DI + d];
    const float dsv = Dvec[k * DI + d];
    float Ac[NN];
#pragma unroll
    for (int n = 0; n < NN; ++n)
        Ac[n] = -__expf(Alog[((size_t)k * DI + d) * NN + n]);
    float h[NN] = {0.f, 0.f, 0.f, 0.f};

    const float* xdb = xdbl + ((size_t)k * MROWS + b * LL) * 32;
    const float* xcb = xc2 + (size_t)b * LL * DI;
    float*       yb  = ybuf + (size_t)b * LL * DI;

    for (int l = 0; l < LL; ++l) {
        if (tid < 32) sh[tid] = xdb[(size_t)l * 32 + tid];
        __syncthreads();
        float dts = dtb;
#pragma unroll
        for (int r = 0; r < RR; ++r) dts += sh[r] * wreg[r];
        const float dtv = (dts > 20.f) ? dts : log1pf(__expf(dts)); // softplus
        const int hw = dir_hw(k, l);
        const float xt = xcb[(size_t)hw * DI + d];
        const float dx = dtv * xt;
        float yv = 0.f;
#pragma unroll
        for (int n = 0; n < NN; ++n) {
            h[n] = __expf(dtv * Ac[n]) * h[n] + dx * sh[RR + n];
            yv  += h[n] * sh[RR + NN + n];
        }
        __hip_atomic_fetch_add(&yb[(size_t)hw * DI + d], yv + xt * dsv,
                               __ATOMIC_RELAXED, __HIP_MEMORY_SCOPE_AGENT);
        __syncthreads();
    }
}

__global__ __launch_bounds__(256) void f32_to_f16_kernel(
    const float* __restrict__ s, _Float16* __restrict__ dst, int n)
{
    const int i = blockIdx.x * 256 + threadIdx.x;
    if (i < n) dst[i] = (_Float16)s[i];
}

// ---------------------------------------------------------------------------
extern "C" void kernel_launch(void* const* d_in, const int* in_sizes, int n_in,
                              void* d_out, int out_size, void* d_ws, size_t ws_size,
                              hipStream_t stream)
{
    const float* x        = (const float*)d_in[0];
    const float* ln1_g    = (const float*)d_in[1];
    const float* ln1_b    = (const float*)d_in[2];
    const float* in_projW = (const float*)d_in[3];
    const float* conv_W   = (const float*)d_in[4];
    const float* conv_b   = (const float*)d_in[5];
    const float* x_projW  = (const float*)d_in[6];
    const float* dt_W     = (const float*)d_in[7];
    const float* dt_b     = (const float*)d_in[8];
    const float* A_log    = (const float*)d_in[9];
    const float* Ds       = (const float*)d_in[10];
    const float* onorm_g  = (const float*)d_in[11];
    const float* onorm_b  = (const float*)d_in[12];
    const float* out_projW= (const float*)d_in[13];
    const float* ln2_g    = (const float*)d_in[14];
    const float* ln2_b    = (const float*)d_in[15];
    const float* fc1_W    = (const float*)d_in[16];
    const float* fc1_b    = (const float*)d_in[17];
    const float* fc2_W    = (const float*)d_in[18];
    const float* fc2_b    = (const float*)d_in[19];
    float* out = (float*)d_out;

    char* ws = (char*)d_ws;
    size_t off = 0;
    auto alloc = [&](size_t bytes) -> char* {
        char* p = ws + off;
        off += (bytes + 255) & ~(size_t)255;
        return p;
    };

    _Float16* xn16   = (_Float16*)alloc((size_t)MROWS * CC * 2);
    _Float16* wIn16  = (_Float16*)alloc((size_t)2 * DI * CC * 2);
    _Float16* wOut16 = (_Float16*)alloc((size_t)CC * DI * 2);
    _Float16* wF116  = (_Float16*)alloc((size_t)HID * CC * 2);
    _Float16* wF216  = (_Float16*)alloc((size_t)CC * HID * 2);
    _Float16* wXp16  = (_Float16*)alloc((size_t)KK * 32 * DI * 2);
    float*    xz     = (float*)   alloc((size_t)MROWS * 2 * DI * 4);
    float*    xc2    = (float*)   alloc((size_t)MROWS * DI * 4);
    _Float16* xs16   = (_Float16*)alloc((size_t)KK * MROWS * DI * 2);
    float*    xdbl   = (float*)   alloc((size_t)KK * MROWS * 32 * 4);
    float*    ybuf   = (float*)   alloc((size_t)MROWS * DI * 4);
    _Float16* y16    = (_Float16*)alloc((size_t)MROWS * DI * 2);
    float*    xr     = (float*)   alloc((size_t)MROWS * CC * 4);
    _Float16* m16    = (_Float16*)alloc((size_t)MROWS * CC * 2);
    _Float16* h16    = (_Float16*)alloc((size_t)MROWS * HID * 2);

    // zero the direction-accumulation buffer (graph-capture safe)
    hipMemsetAsync(ybuf, 0, (size_t)MROWS * DI * 4, stream);

    // weight conversions f32 -> f16
    auto cvt = [&](const float* s, _Float16* d, int n) {
        f32_to_f16_kernel<<<(n + 255) / 256, 256, 0, stream>>>(s, d, n);
    };
    cvt(in_projW,  wIn16,  2 * DI * CC);
    cvt(out_projW, wOut16, CC * DI);
    cvt(fc1_W,     wF116,  HID * CC);
    cvt(fc2_W,     wF216,  CC * HID);
    cvt(x_projW,   wXp16,  KK * 32 * DI);

    // TDM warm of the f16 weight matrices (tensor_load_to_lds -> L2)
    tdm_warm_kernel<<<2 * DI / 64, 32, 0, stream>>>(wIn16, 2 * DI, CC);
    tdm_warm_kernel<<<HID / 64,   32, 0, stream>>>(wF116, HID, CC);
    tdm_warm_kernel<<<CC / 64,    32, 0, stream>>>(wF216, CC, HID);
    tdm_warm_kernel<<<CC / 64,    32, 0, stream>>>(wOut16, CC, DI);

    // 1) LN1 (wave/pixel)
    ln1_kernel<<<MROWS / 8, 256, 0, stream>>>(x, ln1_g, ln1_b, xn16);

    // 2) in_proj GEMM: (8192 x 384) x (384 x 1536) -> xz
    {
        const int waves = (MROWS / 16) * (2 * DI / 64);
        gemm_wmma_kernel<4, GM_PLAIN><<<(waves + 7) / 8, 256, 0, stream>>>(
            xn16, wIn16, xz, nullptr, nullptr, nullptr, MROWS, 2 * DI, CC);
    }

    // 3) depthwise conv + SiLU
    conv_silu_kernel<<<dim3(3, MROWS), 256, 0, stream>>>(xz, conv_W, conv_b, xc2);

    // 4) directional sequences (f16 for WMMA A-matrix)
    xs_build_kernel<<<(int)(((size_t)KK * MROWS * DI) / 256), 256, 0, stream>>>(
        xc2, xs16);

    // 5) x_proj GEMMs per direction: (8192 x 768) x (768 x 32) -> xdbl
    for (int k = 0; k < KK; ++k) {
        const int waves = (MROWS / 16) * (32 / 32);
        gemm_wmma_kernel<2, GM_PLAIN><<<(waves + 7) / 8, 256, 0, stream>>>(
            xs16 + (size_t)k * MROWS * DI, wXp16 + (size_t)k * 32 * DI,
            xdbl + (size_t)k * MROWS * 32, nullptr, nullptr, nullptr,
            MROWS, 32, DI);
    }

    // 6) fused selective scan (+ D skip, + directional recombination)
    scan_kernel<<<dim3(3, KK * BB), 256, 0, stream>>>(
        xdbl, xc2, dt_W, dt_b, A_log, Ds, ybuf);

    // 7) out-norm * silu(z) -> f16
    ynorm_kernel<<<MROWS / 8, 256, 0, stream>>>(ybuf, xz, onorm_g, onorm_b, y16);

    // 8) out_proj GEMM + shortcut(x) -> xr
    {
        const int waves = (MROWS / 16) * (CC / 64);
        gemm_wmma_kernel<4, GM_XRES><<<(waves + 7) / 8, 256, 0, stream>>>(
            y16, wOut16, xr, nullptr, nullptr, x, MROWS, CC, DI);
    }

    // 9) LN2 -> f16
    ln_rows_kernel<<<MROWS / 8, 256, 0, stream>>>(xr, ln2_g, ln2_b, m16);

    // 10) fc1 GEMM + bias + GELU -> f16
    {
        const int waves = (MROWS / 16) * (HID / 64);
        gemm_wmma_kernel<4, GM_GELU><<<(waves + 7) / 8, 256, 0, stream>>>(
            m16, wF116, nullptr, h16, fc1_b, nullptr, MROWS, HID, CC);
    }

    // 11) fc2 GEMM + bias + residual(xr) -> d_out (B,H,W,C)
    {
        const int waves = (MROWS / 16) * (CC / 64);
        gemm_wmma_kernel<4, GM_BIASRES><<<(waves + 7) / 8, 256, 0, stream>>>(
            h16, wF216, out, nullptr, fc2_b, xr, MROWS, CC, HID);
    }
}
